// XCATransformer_16509854286254
// MI455X (gfx1250) — matrix operand order repaired and checked by reading, not compiled, not run on hardware
//
#include <hip/hip_runtime.h>
#include <hip/hip_bf16.h>

// Problem constants (from reference)
#define TOK   36864   // B*H*W tokens
#define SEQ   2304    // tokens per image
#define DDIM  384
#define NHEAD 8
#define DH    48
#define MDIM  1536
#define NB    16
#define HW    48
#define QKVW  1152

typedef __attribute__((ext_vector_type(16))) __bf16 v16bf;
typedef __attribute__((ext_vector_type(8)))  float  v8f;

#define USE_ASYNC_LDS 1

__device__ __forceinline__ float gelu_exact(float x) {
    return 0.5f * x * (1.0f + erff(x * 0.70710678118654752f));
}

#ifdef USE_ASYNC_LDS
// GLOBAL_LOAD_ASYNC_TO_LDS_B128 (GV mode): vdst = per-lane LDS byte address,
// vaddr = per-lane 64-bit global address. Tracked by ASYNCcnt.
__device__ __forceinline__ void async_cp16(const __bf16* g, __bf16* l) {
    unsigned lds = (unsigned)(uintptr_t)l;               // low 32b = LDS offset
    unsigned long long ga = (unsigned long long)(uintptr_t)g;
    asm volatile("global_load_async_to_lds_b128 %0, %1, off"
                 :: "v"(lds), "v"(ga)
                 : "memory");
}
__device__ __forceinline__ void wait_async0() {
    asm volatile("s_wait_asynccnt 0x0" ::: "memory");
}
#endif

// ------------------- weight pre-transpose + bf16 convert --------------------
// out[n*K + k] = (bf16)W[k*N + n]  (so GEMM B-tiles are K-contiguous bf16)
__global__ void wt_transpose_bf16(const float* __restrict__ W, __bf16* __restrict__ out,
                                  int K, int N)
{
    int idx = blockIdx.x * 256 + threadIdx.x;
    int n = idx / K;
    int k = idx % K;
    out[idx] = (__bf16)W[(size_t)k * N + n];
}

// ----------------------------- LayerNorm -----------------------------------
template<bool OUT_F32>
__global__ void ln_kernel(const float* __restrict__ x, const float* __restrict__ g,
                          const float* __restrict__ b, void* __restrict__ out)
{
    int wave = threadIdx.x >> 5;
    int lane = threadIdx.x & 31;
    int tok  = blockIdx.x * 8 + wave;
    const float* row = x + (size_t)tok * DDIM;
    float vals[12];
    float s = 0.f, sq = 0.f;
#pragma unroll
    for (int i = 0; i < 12; ++i) {
        float v = row[lane + i * 32];
        vals[i] = v; s += v; sq += v * v;
    }
#pragma unroll
    for (int m = 16; m >= 1; m >>= 1) {
        s  += __shfl_xor(s, m, 32);
        sq += __shfl_xor(sq, m, 32);
    }
    float mean = s * (1.0f / DDIM);
    float var  = sq * (1.0f / DDIM) - mean * mean;
    float rst  = rsqrtf(var + 1e-5f);
    if (OUT_F32) {
        float* o = (float*)out + (size_t)tok * DDIM;
#pragma unroll
        for (int i = 0; i < 12; ++i) {
            int c = lane + i * 32;
            o[c] = (vals[i] - mean) * rst * g[c] + b[c];
        }
    } else {
        __bf16* o = (__bf16*)out + (size_t)tok * DDIM;
#pragma unroll
        for (int i = 0; i < 12; ++i) {
            int c = lane + i * 32;
            o[c] = (__bf16)((vals[i] - mean) * rst * g[c] + b[c]);
        }
    }
}

// ----------------------------- WMMA GEMM -----------------------------------
// C[M x N] = A(bf16)[M x K] * Wt(bf16)[N x K] (B pre-transposed, K-contig).
// Block tile 128x128, 8 waves in 4x2; wave tile 32x64 -> 8 accumulators,
// 8x v_wmma_f32_16x16x32_bf16 per k-step. LDS double-buffered; tile k+1 is
// fetched with GLOBAL_LOAD_ASYNC_TO_LDS overlapping the WMMAs.
#define BM 128
#define BN 128
#define BK 32
#define AS 48   // padded LDS stride in elements (96B, multiple of 16B)

#define GM_RAW   0   // out bf16 = acc
#define GM_GELU  1   // out bf16 = gelu(acc + bias)
#define GM_RESID 2   // out f32  = (acc + bias)*scale + resid

template<int MODE>
__global__ void gemm_wmma(const __bf16* __restrict__ A, const __bf16* __restrict__ Wt,
                          const float* __restrict__ bias, const float* __restrict__ scale,
                          const float* __restrict__ resid, void* __restrict__ out,
                          int K, int Ncols)
{
    __shared__ __bf16 Alds[2][BM * AS];
    __shared__ __bf16 Blds[2][BN * AS];
    int tid  = threadIdx.x;
    int lane = tid & 31;
    int w    = tid >> 5;
    int wr   = w & 3;    // wave row group: rows [wr*32, +32)
    int wc   = w >> 2;   // wave col group: cols [wc*64, +64)
    int m0 = blockIdx.y * BM;
    int n0 = blockIdx.x * BN;

    // global staging indices: 16B per access, K-contiguous on both A and B
    int gr = tid >> 2;               // 0..63
    int gc = (tid & 3) * 8;          // 0,8,16,24

    const __bf16* Arow0 = A  + (size_t)(m0 + gr)      * K + gc;
    const __bf16* Arow1 = A  + (size_t)(m0 + gr + 64) * K + gc;
    const __bf16* Brow0 = Wt + (size_t)(n0 + gr)      * K + gc;
    const __bf16* Brow1 = Wt + (size_t)(n0 + gr + 64) * K + gc;

    __bf16* La0 = &Alds[0][gr * AS + gc];
    __bf16* La1 = &Alds[0][(gr + 64) * AS + gc];
    __bf16* Lb0 = &Blds[0][gr * AS + gc];
    __bf16* Lb1 = &Blds[0][(gr + 64) * AS + gc];
    const int LBUF = BM * AS;   // element offset between the two LDS buffers

    v8f acc[2][4];
#pragma unroll
    for (int ai = 0; ai < 2; ++ai)
#pragma unroll
        for (int bj = 0; bj < 4; ++bj) acc[ai][bj] = (v8f){};

    int nk = K / BK;

#ifdef USE_ASYNC_LDS
    async_cp16(Arow0, La0);
    async_cp16(Arow1, La1);
    async_cp16(Brow0, Lb0);
    async_cp16(Brow1, Lb1);
    wait_async0();
#else
    *(uint4*)La0 = *(const uint4*)(Arow0);
    *(uint4*)La1 = *(const uint4*)(Arow1);
    *(uint4*)Lb0 = *(const uint4*)(Brow0);
    *(uint4*)Lb1 = *(const uint4*)(Brow1);
#endif
    __syncthreads();

    // fragment addresses (ISA 16-bit A/B VGPR layouts)
    int arow_f = wr * 32 + (lane & 15);
    int koffA  = (lane >> 4) * 8;
    int bcol_f = wc * 64 + (lane & 15);
    int koffB  = (lane >> 4) * 8;

    for (int kt = 0; kt < nk; ++kt) {
        int cur = kt & 1, nxt = cur ^ 1;
#ifdef USE_ASYNC_LDS
        if (kt + 1 < nk) {
            int ko = (kt + 1) * BK;
            async_cp16(Arow0 + ko, La0 + nxt * LBUF);
            async_cp16(Arow1 + ko, La1 + nxt * LBUF);
            async_cp16(Brow0 + ko, Lb0 + nxt * LBUF);
            async_cp16(Brow1 + ko, Lb1 + nxt * LBUF);
        }
#else
        uint4 ra0, ra1, rb0, rb1;
        if (kt + 1 < nk) {
            int ko = (kt + 1) * BK;
            ra0 = *(const uint4*)(Arow0 + ko);
            ra1 = *(const uint4*)(Arow1 + ko);
            rb0 = *(const uint4*)(Brow0 + ko);
            rb1 = *(const uint4*)(Brow1 + ko);
        }
        if (kt + 2 < nk) {
            __builtin_prefetch(Arow0 + (kt + 2) * BK, 0, 1);
            __builtin_prefetch(Brow0 + (kt + 2) * BK, 0, 1);
        }
#endif

        union { v16bf v; uint4 u[2]; } fa[2], fb[4];
#pragma unroll
        for (int ai = 0; ai < 2; ++ai) {
            const __bf16* p = &Alds[cur][(arow_f + ai * 16) * AS + koffA];
            fa[ai].u[0] = *(const uint4*)(p);
            fa[ai].u[1] = *(const uint4*)(p + 16);
        }
#pragma unroll
        for (int bj = 0; bj < 4; ++bj) {
            const __bf16* p = &Blds[cur][(bcol_f + bj * 16) * AS + koffB];
            fb[bj].u[0] = *(const uint4*)(p);
            fb[bj].u[1] = *(const uint4*)(p + 16);
        }
#pragma unroll
        for (int ai = 0; ai < 2; ++ai)
#pragma unroll
            for (int bj = 0; bj < 4; ++bj)
                acc[ai][bj] = __builtin_amdgcn_wmma_f32_16x16x32_bf16(
                    false, fa[ai].v, false, fb[bj].v, (short)0, acc[ai][bj], false, false);

#ifdef USE_ASYNC_LDS
        if (kt + 1 < nk) wait_async0();
#else
        if (kt + 1 < nk) {
            *(uint4*)(La0 + nxt * LBUF) = ra0;
            *(uint4*)(La1 + nxt * LBUF) = ra1;
            *(uint4*)(Lb0 + nxt * LBUF) = rb0;
            *(uint4*)(Lb1 + nxt * LBUF) = rb1;
        }
#endif
        __syncthreads();
    }

    // Epilogue. C/D layout: lane L col=L%16, VGPR r -> row r + 8*(L>=16).
    int rsub = (lane >> 4) * 8;
    int cl   = lane & 15;
#pragma unroll
    for (int ai = 0; ai < 2; ++ai) {
#pragma unroll
        for (int bj = 0; bj < 4; ++bj) {
            int gcol = n0 + wc * 64 + bj * 16 + cl;
#pragma unroll
            for (int r = 0; r < 8; ++r) {
                size_t grow = (size_t)(m0 + wr * 32 + ai * 16 + rsub + r);
                float v = acc[ai][bj][r];
                if (MODE == GM_RAW) {
                    ((__bf16*)out)[grow * Ncols + gcol] = (__bf16)v;
                } else if (MODE == GM_GELU) {
                    v += bias[gcol];
                    ((__bf16*)out)[grow * Ncols + gcol] = (__bf16)gelu_exact(v);
                } else {
                    v += bias[gcol];
                    ((float*)out)[grow * Ncols + gcol] =
                        v * scale[gcol] + resid[grow * Ncols + gcol];
                }
            }
        }
    }
}

// --------------------------- XCA attention ---------------------------------
__global__ void xca_attn_kernel(const __bf16* __restrict__ qkv,
                                const float* __restrict__ temp,
                                __bf16* __restrict__ outb)
{
    __shared__ __bf16 qt[DH][260];
    __shared__ __bf16 kt[DH][260];
    __shared__ float simsh[DH][DH];
    __shared__ float nqs[DH], nks[DH];

    int tid = threadIdx.x;
    int b = blockIdx.x >> 3;
    int h = blockIdx.x & 7;
    size_t tb = (size_t)b * SEQ;
    int qb = h * DH, kb = DDIM + h * DH, vb = 2 * DDIM + h * DH;

    float sacc[9]; int sd[9], se[9];
#pragma unroll
    for (int p = 0; p < 9; ++p) {
        int i = tid + p * 256;
        sd[p] = i / DH; se[p] = i % DH; sacc[p] = 0.f;
    }
    float nacc = 0.f;

    for (int tt = 0; tt < 9; ++tt) {
        for (int li = tid; li < DH * 256; li += 256) {
            int toff = li / DH, r = li % DH;
            size_t rowbase = (tb + tt * 256 + toff) * QKVW;
            qt[r][toff] = qkv[rowbase + qb + r];
            kt[r][toff] = qkv[rowbase + kb + r];
        }
        __syncthreads();
        if (tid < DH) {
            for (int c = 0; c < 256; ++c) { float f = (float)qt[tid][c]; nacc += f * f; }
        } else if (tid < 2 * DH) {
            for (int c = 0; c < 256; ++c) { float f = (float)kt[tid - DH][c]; nacc += f * f; }
        }
#pragma unroll
        for (int p = 0; p < 9; ++p) {
            const __bf16* qr = qt[sd[p]];
            const __bf16* kr = kt[se[p]];
            float a = 0.f;
            for (int c = 0; c < 256; ++c) a += (float)qr[c] * (float)kr[c];
            sacc[p] += a;
        }
        __syncthreads();
    }
    if (tid < DH) nqs[tid] = nacc;
    else if (tid < 2 * DH) nks[tid - DH] = nacc;
    __syncthreads();

    float ts = expf(temp[h]);
#pragma unroll
    for (int p = 0; p < 9; ++p) {
        float nq = fmaxf(sqrtf(nqs[sd[p]]), 1e-12f);
        float nk = fmaxf(sqrtf(nks[se[p]]), 1e-12f);
        simsh[sd[p]][se[p]] = sacc[p] / (nq * nk) * ts;
    }
    __syncthreads();

    int wave = tid >> 5, lane = tid & 31;
    for (int d = wave; d < DH; d += 8) {
        float v0 = simsh[d][lane];
        float v1 = (lane < 16) ? simsh[d][lane + 32] : -3.4e38f;
        float mx = fmaxf(v0, v1);
        for (int m = 16; m >= 1; m >>= 1) mx = fmaxf(mx, __shfl_xor(mx, m, 32));
        float e0 = expf(v0 - mx);
        float e1 = (lane < 16) ? expf(v1 - mx) : 0.f;
        float sm = e0 + e1;
        for (int m = 16; m >= 1; m >>= 1) sm += __shfl_xor(sm, m, 32);
        float inv = 1.0f / sm;
        simsh[d][lane] = e0 * inv;
        if (lane < 16) simsh[d][lane + 32] = e1 * inv;
    }
    __syncthreads();

    for (int t = tid; t < SEQ; t += 256) {
        const __bf16* vrow = qkv + (tb + t) * QKVW + vb;
        float vv[DH];
#pragma unroll
        for (int e = 0; e < DH; ++e) vv[e] = (float)vrow[e];
        __bf16* orow = outb + (tb + t) * DDIM + h * DH;
        for (int d = 0; d < DH; ++d) {
            float a = 0.f;
#pragma unroll
            for (int e = 0; e < DH; ++e) a += simsh[d][e] * vv[e];
            orow[d] = (__bf16)a;
        }
    }
}

// --------------------------- LPI kernels -----------------------------------
__global__ void zero_stats(float* stats) {
    int i = blockIdx.x * 256 + threadIdx.x;
    if (i < 2 * DDIM) stats[i] = 0.f;
}

__global__ void conv_dw_stats(const float* __restrict__ hin, const float* __restrict__ w,
                              const float* __restrict__ bias, float* __restrict__ hout,
                              float* __restrict__ stats)
{
    int idx  = blockIdx.x * 256 + threadIdx.x;
    int c    = idx % DDIM;
    int tokn = idx / DDIM;
    int bimg = tokn / SEQ;
    int rem  = tokn % SEQ;
    int y = rem / HW, x = rem % HW;
    float acc = bias[c];
    const float* wp = w + c * 9;
#pragma unroll
    for (int dy = -1; dy <= 1; ++dy) {
        int yy = y + dy;
        if (yy < 0 || yy >= HW) continue;
#pragma unroll
        for (int dx = -1; dx <= 1; ++dx) {
            int xc = x + dx;
            if (xc < 0 || xc >= HW) continue;
            acc += wp[(dy + 1) * 3 + (dx + 1)] *
                   hin[((size_t)bimg * SEQ + yy * HW + xc) * DDIM + c];
        }
    }
    hout[idx] = acc;
    atomicAdd(&stats[c], acc);
    atomicAdd(&stats[DDIM + c], acc * acc);
}

__global__ void bn_gelu_kernel(const float* __restrict__ h1, const float* __restrict__ stats,
                               const float* __restrict__ g, const float* __restrict__ b,
                               float* __restrict__ hout)
{
    int idx = blockIdx.x * 256 + threadIdx.x;
    int c = idx % DDIM;
    const float invN = 1.0f / (float)TOK;
    float mean = stats[c] * invN;
    float var  = stats[DDIM + c] * invN - mean * mean;
    float v = (h1[idx] - mean) * rsqrtf(var + 1e-5f) * g[c] + b[c];
    hout[idx] = gelu_exact(v);
}

__global__ void conv_dw_resid(const float* __restrict__ hin, const float* __restrict__ w,
                              const float* __restrict__ bias, const float* __restrict__ scale,
                              float* __restrict__ xio)
{
    int idx  = blockIdx.x * 256 + threadIdx.x;
    int c    = idx % DDIM;
    int tokn = idx / DDIM;
    int bimg = tokn / SEQ;
    int rem  = tokn % SEQ;
    int y = rem / HW, x = rem % HW;
    float acc = bias[c];
    const float* wp = w + c * 9;
#pragma unroll
    for (int dy = -1; dy <= 1; ++dy) {
        int yy = y + dy;
        if (yy < 0 || yy >= HW) continue;
#pragma unroll
        for (int dx = -1; dx <= 1; ++dx) {
            int xc = x + dx;
            if (xc < 0 || xc >= HW) continue;
            acc += wp[(dy + 1) * 3 + (dx + 1)] *
                   hin[((size_t)bimg * SEQ + yy * HW + xc) * DDIM + c];
        }
    }
    xio[idx] = xio[idx] + scale[c] * acc;
}

// ------------------------------ driver --------------------------------------
extern "C" void kernel_launch(void* const* d_in, const int* in_sizes, int n_in,
                              void* d_out, int out_size, void* d_ws, size_t ws_size,
                              hipStream_t stream)
{
    (void)in_sizes; (void)n_in; (void)out_size; (void)ws_size;
    const float* xin   = (const float*)d_in[0];
    const float* ln1g  = (const float*)d_in[1];
    const float* ln1b  = (const float*)d_in[2];
    const float* wqkv  = (const float*)d_in[3];
    const float* temp  = (const float*)d_in[4];
    const float* wout  = (const float*)d_in[5];
    const float* bout  = (const float*)d_in[6];
    const float* sattn = (const float*)d_in[7];
    const float* lpig  = (const float*)d_in[8];
    const float* lpib  = (const float*)d_in[9];
    const float* c1w   = (const float*)d_in[10];
    const float* c1b   = (const float*)d_in[11];
    const float* bng   = (const float*)d_in[12];
    const float* bnb   = (const float*)d_in[13];
    const float* c2w   = (const float*)d_in[14];
    const float* c2b   = (const float*)d_in[15];
    const float* slpi  = (const float*)d_in[16];
    const float* ffg   = (const float*)d_in[17];
    const float* ffb   = (const float*)d_in[18];
    const float* w1    = (const float*)d_in[19];
    const float* b1    = (const float*)d_in[20];
    const float* w2    = (const float*)d_in[21];
    const float* b2    = (const float*)d_in[22];
    const float* sff   = (const float*)d_in[23];

    float* x = (float*)d_out;                        // running activations
    char* ws = (char*)d_ws;
    __bf16* r1 = (__bf16*)ws;                        // TOK*DDIM bf16 (LN / attn out)
    char* ws2 = ws + (size_t)TOK * DDIM * 2;
    __bf16* r2b  = (__bf16*)ws2;                     // TOK*MDIM bf16 (qkv / ff1 out)
    float*  r3   = (float*)ws2;                      // TOK*DDIM f32 (LPI, aliases r2b)
    float*  r4   = (float*)(ws2 + (size_t)TOK * DDIM * 4);
    float*  stats = (float*)(ws2 + (size_t)TOK * MDIM * 2);
    __bf16* wtb   = (__bf16*)(ws2 + (size_t)TOK * MDIM * 2 + 4096);

    // transposed bf16 weight regions
    const size_t SZ_QKV = (size_t)QKVW * DDIM;   // per layer
    const size_t SZ_OUT = (size_t)DDIM * DDIM;
    const size_t SZ_W1  = (size_t)MDIM * DDIM;
    const size_t SZ_W2  = (size_t)DDIM * MDIM;
    __bf16* qkvT = wtb;
    __bf16* outT = qkvT + 4 * SZ_QKV;
    __bf16* w1T  = outT + 4 * SZ_OUT;
    __bf16* w2T  = w1T  + 4 * SZ_W1;

    (void)hipMemcpyAsync(x, xin, (size_t)TOK * DDIM * sizeof(float),
                         hipMemcpyDeviceToDevice, stream);

    for (int i = 0; i < 4; ++i) {
        wt_transpose_bf16<<<SZ_QKV / 256, 256, 0, stream>>>(wqkv + i * SZ_QKV,
                                                            qkvT + i * SZ_QKV, DDIM, QKVW);
        wt_transpose_bf16<<<SZ_OUT / 256, 256, 0, stream>>>(wout + i * SZ_OUT,
                                                            outT + i * SZ_OUT, DDIM, DDIM);
        wt_transpose_bf16<<<SZ_W1 / 256, 256, 0, stream>>>(w1 + i * SZ_W1,
                                                           w1T + i * SZ_W1, DDIM, MDIM);
        wt_transpose_bf16<<<SZ_W2 / 256, 256, 0, stream>>>(w2 + i * SZ_W2,
                                                           w2T + i * SZ_W2, MDIM, DDIM);
    }

    const int EW = TOK * DDIM / 256;   // elementwise grid
    dim3 gq(QKVW / BN, TOK / BM);
    dim3 gp(DDIM / BN, TOK / BM);
    dim3 g1(MDIM / BN, TOK / BM);

    for (int i = 0; i < 4; ++i) {
        // ---- XCA ----
        ln_kernel<false><<<TOK / 8, 256, 0, stream>>>(x, ln1g + i * DDIM, ln1b + i * DDIM, r1);
        gemm_wmma<GM_RAW><<<gq, 256, 0, stream>>>(r1, qkvT + i * SZ_QKV,
                                                  nullptr, nullptr, nullptr, r2b, DDIM, QKVW);
        xca_attn_kernel<<<NB * NHEAD, 256, 0, stream>>>(r2b, temp + i * NHEAD, r1);
        gemm_wmma<GM_RESID><<<gp, 256, 0, stream>>>(r1, outT + i * SZ_OUT,
                                                    bout + i * DDIM, sattn + i * DDIM,
                                                    x, x, DDIM, DDIM);
        // ---- LPI ----
        ln_kernel<true><<<TOK / 8, 256, 0, stream>>>(x, lpig + i * DDIM, lpib + i * DDIM, r3);
        zero_stats<<<3, 256, 0, stream>>>(stats);
        conv_dw_stats<<<EW, 256, 0, stream>>>(r3, c1w + (size_t)i * DDIM * 9,
                                              c1b + i * DDIM, r4, stats);
        bn_gelu_kernel<<<EW, 256, 0, stream>>>(r4, stats, bng + i * DDIM, bnb + i * DDIM, r3);
        conv_dw_resid<<<EW, 256, 0, stream>>>(r3, c2w + (size_t)i * DDIM * 9,
                                              c2b + i * DDIM, slpi + i * DDIM, x);
        // ---- FF ----
        ln_kernel<false><<<TOK / 8, 256, 0, stream>>>(x, ffg + i * DDIM, ffb + i * DDIM, r1);
        gemm_wmma<GM_GELU><<<g1, 256, 0, stream>>>(r1, w1T + i * SZ_W1,
                                                   b1 + i * MDIM, nullptr, nullptr,
                                                   r2b, DDIM, MDIM);
        gemm_wmma<GM_RESID><<<gp, 256, 0, stream>>>(r2b, w2T + i * SZ_W2,
                                                    b2 + i * DDIM, sff + i * DDIM,
                                                    x, x, MDIM, DDIM);
    }
}
